// ScaledDotProductAttention_53034256171721
// MI455X (gfx1250) — matrix operand order repaired and checked by reading, not compiled, not run on hardware
//
#include <hip/hip_runtime.h>

typedef __attribute__((ext_vector_type(16))) _Float16 v16h;
typedef __attribute__((ext_vector_type(8)))  _Float16 v8h;
typedef __attribute__((ext_vector_type(4)))  _Float16 v4h;
typedef __attribute__((ext_vector_type(8)))  float    v8f;

#define BH   32
#define SEQ  2048
#define HD   64
#define QK_SCALE 0.125f   // 1/sqrt(64)
#define NEGV (-1e9f)

// ---------------------------------------------------------------------------
// Pre-pass 1: V (BH,S,64) f32  ->  Vt (BH,64,S) f16   (LDS-tiled transpose)
// ---------------------------------------------------------------------------
__global__ __launch_bounds__(256) void vt_transpose_kernel(
    const float* __restrict__ V, _Float16* __restrict__ Vt) {
  __shared__ __align__(16) _Float16 tile[HD][72];  // pad: 144B row stride, 16B-aligned
  const int s0 = blockIdx.x * 64;
  const int b  = blockIdx.y;
  const float* vb = V + ((size_t)b * SEQ + s0) * HD;
  const int t = threadIdx.x;
#pragma unroll
  for (int k = 0; k < 16; ++k) {
    int n  = t + k * 256;      // 0..4095 over 64 rows x 64 dims
    int sl = n >> 6;
    int d  = n & 63;
    tile[d][sl] = (_Float16)vb[(size_t)sl * HD + d];
  }
  __syncthreads();
  const int d    = t >> 2;
  const int base = (t & 3) * 16;
  const uint4* src = (const uint4*)&tile[d][base];
  uint4* dst = (uint4*)(Vt + ((size_t)b * HD + d) * SEQ + s0 + base);
  dst[0] = src[0];
  dst[1] = src[1];
}

// ---------------------------------------------------------------------------
// Pre-pass 2: K f32 -> f16 (elementwise), so the hot loop does b128 loads
// ---------------------------------------------------------------------------
__global__ __launch_bounds__(256) void f32_to_f16_kernel(
    const float* __restrict__ X, _Float16* __restrict__ Y, int n4) {
  int i = blockIdx.x * 256 + threadIdx.x;
  if (i < n4) {
    float4 f = ((const float4*)X)[i];
    v4h h;
    h[0] = (_Float16)f.x; h[1] = (_Float16)f.y;
    h[2] = (_Float16)f.z; h[3] = (_Float16)f.w;
    ((v4h*)Y)[i] = h;
  }
}

// ---------------------------------------------------------------------------
// Flash attention main kernel.
// Block = 128 threads = 4 waves; wave w owns q rows [qb + w*16, qb + w*16 + 16).
// Score tiles are computed TRANSPOSED:  T = K_tile(16x64) x Q^T  so that
//  (a) A (K rows) and B (Q rows) fragments are contiguous row-major loads, and
//  (b) T's D-layout is bit-identical to the A-fragment layout needed for P*V
//      after f32->f16 pair packing (no cross-lane data movement).
// Mask (134 MB, zero reuse) is streamed with NT hints so the reusable Kh/Vt
// working set (16 MB) stays resident in the 192 MB L2.
// ---------------------------------------------------------------------------
__global__ __launch_bounds__(128) void fa_fwd_kernel(
    const float*        __restrict__ Q,
    const _Float16*     __restrict__ Kh,
    const unsigned char* __restrict__ Mask,
    const _Float16*     __restrict__ Vt,
    float*              __restrict__ Out) {
  const int lane = threadIdx.x & 31;
  const int wave = threadIdx.x >> 5;
  const int hi   = lane >> 4;        // 0 = lanes 0-15, 1 = lanes 16-31
  const int ln   = lane & 15;
  const int b    = blockIdx.y;
  const int qb   = blockIdx.x * 64 + wave * 16;
  const int qrow = qb + ln;

  // B fragments of Q^T (pre-scaled). B-layout: lane n(<16) holds K(d)=0..15
  // contiguous; lane n+16 holds d=16..31. One 16-element contiguous chunk/lane.
  v16h bq[2];
#pragma unroll
  for (int f = 0; f < 2; ++f) {
    const float* qp = Q + ((size_t)b * SEQ + qrow) * HD + f * 32 + hi * 16;
#pragma unroll
    for (int j = 0; j < 4; ++j) {
      float4 fv = *(const float4*)(qp + 4 * j);
      bq[f][4 * j + 0] = (_Float16)(fv.x * QK_SCALE);
      bq[f][4 * j + 1] = (_Float16)(fv.y * QK_SCALE);
      bq[f][4 * j + 2] = (_Float16)(fv.z * QK_SCALE);
      bq[f][4 * j + 3] = (_Float16)(fv.w * QK_SCALE);
    }
  }

  v8f o[4] = {{0,0,0,0,0,0,0,0},{0,0,0,0,0,0,0,0},
              {0,0,0,0,0,0,0,0},{0,0,0,0,0,0,0,0}};
  float m = -3.0e38f;
  float l = 0.0f;

  const unsigned char* mrow = Mask + ((size_t)b * SEQ + qrow) * SEQ;
  const _Float16* krows = Kh + (size_t)b * SEQ * HD;
  const _Float16* vtb   = Vt + (size_t)b * HD * SEQ;

  for (int kv = 0; kv < SEQ; kv += 32) {
    // ---- T[sub] = K_tile(16 kv x 64 d) @ Q^T  (two k-dim chunks of 32)
    v8f T[2];
#pragma unroll
    for (int sub = 0; sub < 2; ++sub) {
      v8f acc = {0,0,0,0,0,0,0,0};
      const _Float16* kp = krows + (size_t)(kv + sub * 16 + ln) * HD;
#pragma unroll
      for (int f = 0; f < 2; ++f) {
        // A-layout: lane m(<16): d = f*32 + {0..15}; lane m+16: +16
        v8h c0 = *(const v8h*)(kp + f * 32 + hi * 16);
        v8h c1 = *(const v8h*)(kp + f * 32 + hi * 16 + 8);
        v16h ak = __builtin_shufflevector(c0, c1, 0,1,2,3,4,5,6,7,
                                                   8,9,10,11,12,13,14,15);
        acc = __builtin_amdgcn_wmma_f32_16x16x32_f16(
            false, ak, false, bq[f], (short)0, acc, false, false);
      }
      T[sub] = acc;
    }

    // ---- mask + online softmax. In T's D-layout this lane holds, for row
    // q=ln, kv offsets {sub*16 + hi*8 + r : r=0..7, sub=0..1}; the partner
    // lane (lane ^ 16) holds the other 16 kv values of the same q row.
    float p[16];
    float tmax = -3.0e38f;
#pragma unroll
    for (int sub = 0; sub < 2; ++sub) {
      unsigned long long mb = __builtin_nontemporal_load(
          (const unsigned long long*)(mrow + kv + sub * 16 + hi * 8));
#pragma unroll
      for (int r = 0; r < 8; ++r) {
        unsigned byte = (unsigned)(mb >> (8 * r)) & 0xffu;
        float s = byte ? NEGV : T[sub][r];
        p[sub * 8 + r] = s;
        tmax = fmaxf(tmax, s);
      }
    }
    tmax = fmaxf(tmax, __shfl_xor(tmax, 16, 32));
    const float mnew  = fmaxf(m, tmax);
    const float alpha = __expf(m - mnew);
    float psum = 0.0f;
#pragma unroll
    for (int i = 0; i < 16; ++i) { p[i] = __expf(p[i] - mnew); psum += p[i]; }
    psum += __shfl_xor(psum, 16, 32);
    l = l * alpha + psum;
    m = mnew;

    // ---- rescale O accumulators: C/D-layout row q = r + hi*8 lives in VGPR r
#pragma unroll
    for (int r = 0; r < 8; ++r) {
      float ar = __shfl(alpha, r + hi * 8, 32);
#pragma unroll
      for (int t = 0; t < 4; ++t) o[t][r] *= ar;
    }

    // ---- pack P: T's D-layout == A-fragment layout after pairwise packing
    v16h ap;
#pragma unroll
    for (int i = 0; i < 16; ++i) ap[i] = (_Float16)p[i];

    // ---- O += P(16x32) @ V(32x64); B from pre-transposed Vt: lane n(=d)
    // holds kv {0..7, 16..23} (+8 for high lanes) from the Vt row.
#pragma unroll
    for (int t = 0; t < 4; ++t) {
      const _Float16* vp = vtb + (size_t)(t * 16 + ln) * SEQ + kv + hi * 8;
      v16h bv = __builtin_shufflevector(*(const v8h*)vp, *(const v8h*)(vp + 16),
                                        0,1,2,3,4,5,6,7,8,9,10,11,12,13,14,15);
      o[t] = __builtin_amdgcn_wmma_f32_16x16x32_f16(
          false, ap, false, bv, (short)0, o[t], false, false);
    }
  }

  // ---- epilogue: normalize by row sum and store (f32 output, streaming)
  const float rl = 1.0f / l;
#pragma unroll
  for (int r = 0; r < 8; ++r) {
    const float rlr = __shfl(rl, r + hi * 8, 32);
    const int q = qb + r + hi * 8;
#pragma unroll
    for (int t = 0; t < 4; ++t) {
      const int d = t * 16 + ln;
      __builtin_nontemporal_store(o[t][r] * rlr,
                                  Out + ((size_t)b * SEQ + q) * HD + d);
    }
  }
}

// ---------------------------------------------------------------------------
extern "C" void kernel_launch(void* const* d_in, const int* in_sizes, int n_in,
                              void* d_out, int out_size, void* d_ws, size_t ws_size,
                              hipStream_t stream) {
  const float*         q    = (const float*)d_in[0];
  const float*         k    = (const float*)d_in[1];
  const float*         v    = (const float*)d_in[2];
  const unsigned char* mask = (const unsigned char*)d_in[3];
  float* out = (float*)d_out;

  _Float16* Vt = (_Float16*)d_ws;                                      // 8 MB
  _Float16* Kh = (_Float16*)((char*)d_ws + (size_t)BH * HD * SEQ * 2); // 8 MB

  vt_transpose_kernel<<<dim3(SEQ / 64, BH), 256, 0, stream>>>(v, Vt);

  const int n4 = BH * SEQ * HD / 4;
  f32_to_f16_kernel<<<dim3((n4 + 255) / 256), 256, 0, stream>>>(k, Kh, n4);

  fa_fwd_kernel<<<dim3(SEQ / 64, BH), 128, 0, stream>>>(q, Kh, mask, Vt, out);
}
